// LSTMModel_73658689126540
// MI455X (gfx1250) — matrix operand order repaired and checked by reading, not compiled, not run on hardware
//
#include <hip/hip_runtime.h>

#define B_TOT    256
#define T_STEPS  1024
#define M_OUT    64
#define K_HID    512
#define G_TOT    2048      // 4*K gate width
#define IN_F     65        // M+1
#define IN_PAD   96        // x padded to 3 k-tiles of 32
#define K_CAT    608       // IN_PAD + K_HID combined A width
#define KT_CAT   19        // 608 / 32
#define KT_HID   16        // 512 / 32
#define BTILE    16
#define A_STRIDE 616       // K_CAT + 8 halves (bank-conflict pad)

typedef _Float16 half_t;
typedef _Float16     v16h  __attribute__((ext_vector_type(16)));
typedef float        f32x8 __attribute__((ext_vector_type(8)));
typedef unsigned int u32x4 __attribute__((ext_vector_type(4)));

union Frag { v16h v; u32x4 q[2]; };

__device__ __forceinline__ float sigm(float x) { return 1.0f / (1.0f + __expf(-x)); }
// tanh(x) = 1 - 2/(e^{2x}+1); exact limits at +/-inf, cheap on VALU.
__device__ __forceinline__ float tanh_fast(float x) { return 1.0f - 2.0f / (__expf(2.0f * x) + 1.0f); }

// 16x32 f16 A-fragment from LDS, contiguous k order (same as load_b):
//  lane<16 : M=lane,    K = {k0..k0+15}
//  lane>=16: M=lane-16, K = {k0+16..k0+31}
__device__ __forceinline__ Frag load_a(const half_t* base, int k0, int lane) {
  const int m = lane & 15, hf = lane >> 4;
  const half_t* p = base + m * A_STRIDE + k0 + hf * 16;
  Frag f;
  f.q[0] = *(const u32x4*)(p);
  f.q[1] = *(const u32x4*)(p + 8);
  return f;
}

// B fragments pre-packed in fragment order: per tile, 32 lanes x 16 halves;
// lane l holds B[k = kt*32 + (l>>4)*16 + j][n = nt*16 + (l&15)], j = 0..15.
__device__ __forceinline__ Frag load_b(const half_t* packed, int tile, int lane) {
  const half_t* p = packed + ((size_t)tile << 9) + (lane << 4);
  Frag f;
  f.q[0] = *(const u32x4*)(p);
  f.q[1] = *(const u32x4*)(p + 8);
  return f;
}

// Pack the concatenated gate weight B[k][n], k in [0,608):
//   k <  65 : W_ih[n][k]      (W_ih row-major [2048][65])
//   k <  96 : 0               (padding)
//   k >= 96 : W_hh[n][k-96]   (W_hh row-major [2048][512])
__global__ void pack_wcat_kernel(const float* __restrict__ W_ih, const float* __restrict__ W_hh,
                                 half_t* __restrict__ dst) {
  const int per_nt = KT_CAT * 512;
  const int total = G_TOT * K_CAT;
  for (int e = blockIdx.x * blockDim.x + threadIdx.x; e < total; e += gridDim.x * blockDim.x) {
    int nt  = e / per_nt;
    int rem = e - nt * per_nt;
    int kt  = rem >> 9;
    int li  = rem & 511;
    int l = li >> 4, j = li & 15;
    int n = nt * 16 + (l & 15);
    int k = kt * 32 + ((l >> 4) << 4) + j;
    float v = 0.0f;
    if (k < IN_F)         v = W_ih[(size_t)n * IN_F + k];
    else if (k >= IN_PAD) v = W_hh[(size_t)n * K_HID + (k - IN_PAD)];
    dst[e] = (half_t)v;
  }
}

// Pack V_w ([64][512] row-major) as B[k][n] = V_w[n][k], 4 nt x 16 kt.
__global__ void pack_vw_kernel(const float* __restrict__ V_w, half_t* __restrict__ dst) {
  const int per_nt = KT_HID * 512;
  const int total = M_OUT * K_HID;
  for (int e = blockIdx.x * blockDim.x + threadIdx.x; e < total; e += gridDim.x * blockDim.x) {
    int nt  = e / per_nt;
    int rem = e - nt * per_nt;
    int kt  = rem >> 9;
    int li  = rem & 511;
    int l = li >> 4, j = li & 15;
    int n = nt * 16 + (l & 15);
    int k = kt * 32 + ((l >> 4) << 4) + j;
    dst[e] = (half_t)V_w[(size_t)n * K_HID + k];
  }
}

// Output projection of one 16x16 tile: out[b0..+15][t][wave*16..+15] = h @ V_w^T + V_b.
// hsrc points at the h-region (column 96) of the combined A buffer.
__device__ __forceinline__ void project_step(const half_t* __restrict__ hsrc,
                                             const half_t* __restrict__ Vw_p,
                                             float* __restrict__ out,
                                             int b0, int tstep, int wave, int lane, float vbl) {
  const int m = lane & 15, hf = lane >> 4;
  f32x8 acc;
#pragma unroll
  for (int r = 0; r < 8; ++r) acc[r] = vbl;
#pragma unroll 1
  for (int kt = 0; kt < KT_HID; ++kt) {
    Frag a = load_a(hsrc, kt * 32, lane);
    Frag b = load_b(Vw_p, wave * KT_HID + kt, lane);
    acc = __builtin_amdgcn_wmma_f32_16x16x32_f16(false, a.v, false, b.v, (short)0, acc, false, false);
  }
#pragma unroll
  for (int r = 0; r < 8; ++r)
    out[((size_t)(b0 + r + 8 * hf) * T_STEPS + tstep) * M_OUT + wave * 16 + m] = acc[r];
}

__global__ void __launch_bounds__(1024, 1)
lstm_fused_kernel(const float* __restrict__ inp,      // [256][1024][65]
                  const float* __restrict__ b_ih,     // [2048]
                  const float* __restrict__ b_hh,     // [2048]
                  const float* __restrict__ vbias,    // [64]
                  const half_t* __restrict__ Wcat_p,  // packed [128 nt][19 kt][512]
                  const half_t* __restrict__ Vw_p,    // packed [4 nt][16 kt][512]
                  float* __restrict__ out)            // [256][1024][64]
{
  extern __shared__ char smem[];
  half_t* ab0 = (half_t*)smem;                        // combined [x | h] buffer, double
  half_t* ab1 = ab0 + BTILE * A_STRIDE;
  float*  bias_s = (float*)(ab1 + BTILE * A_STRIDE);  // b_ih + b_hh
  float*  vb_s   = bias_s + G_TOT;

  const int tid  = threadIdx.x;
  const int lane = tid & 31;
  const int wave = tid >> 5;                          // 0..31
  const int b0   = blockIdx.x * BTILE;
  const int m    = lane & 15;
  const int hf   = lane >> 4;
  const int cb   = wave * 16;                         // this wave's 16 cell columns

  for (int i = tid; i < G_TOT; i += 1024) bias_s[i] = b_ih[i] + b_hh[i];
  if (tid < M_OUT) vb_s[tid] = vbias[tid];
  for (int i = tid; i < 2 * BTILE * A_STRIDE; i += 1024) ab0[i] = (half_t)0.0f; // h_{-1}=0, x pads=0
  __syncthreads();

  float gb[4];
#pragma unroll
  for (int g = 0; g < 4; ++g) gb[g] = bias_s[g * K_HID + cb + m];
  const float vbl = (wave < 4) ? vb_s[wave * 16 + m] : 0.0f;

  float creg[8];                                      // cell state, register-resident
#pragma unroll
  for (int r = 0; r < 8; ++r) creg[r] = 0.0f;

#pragma unroll 1
  for (int t = 0; t < T_STEPS; ++t) {
    const int cur = t & 1;
    const half_t* aprev = cur ? ab0 : ab1;            // holds x_t (after phase A) and h_{t-1}
    half_t* anew        = cur ? ab1 : ab0;            // h_t written into its h-region

    // ---- Phase A: stage x_t into aprev's x-region; waves 0..3 project h_{t-1}. ----
    for (int i = tid; i < BTILE * IN_F; i += 1024) {
      int r = i / IN_F, cc = i - r * IN_F;
      ((half_t*)aprev)[r * A_STRIDE + cc] =
          (half_t)inp[((size_t)(b0 + r) * T_STEPS + t) * IN_F + cc];
    }
    if (wave < 4 && t > 0) project_step(aprev + IN_PAD, Vw_p, out, b0, t - 1, wave, lane, vbl);
    __syncthreads();

    // ---- Phase B: gates = bias + [x_t | h_{t-1}] @ Wcat^T, one 19-ktile WMMA loop. ----
    f32x8 acc0, acc1, acc2, acc3;
#pragma unroll
    for (int r = 0; r < 8; ++r) { acc0[r] = gb[0]; acc1[r] = gb[1]; acc2[r] = gb[2]; acc3[r] = gb[3]; }

    const int tb = wave * KT_CAT;                     // tile base for gate 0
#pragma unroll 1
    for (int kt = 0; kt < KT_CAT; ++kt) {
      // Issue A (LDS) and all four B (global) loads before any WMMA so the
      // later WMMAs only need partial loadcnt waits.
      Frag a  = load_a(aprev, kt * 32, lane);
      Frag b0f = load_b(Wcat_p, tb + kt, lane);
      Frag b1f = load_b(Wcat_p, tb + 32 * KT_CAT + kt, lane);
      Frag b2f = load_b(Wcat_p, tb + 64 * KT_CAT + kt, lane);
      Frag b3f = load_b(Wcat_p, tb + 96 * KT_CAT + kt, lane);
      acc0 = __builtin_amdgcn_wmma_f32_16x16x32_f16(false, a.v, false, b0f.v, (short)0, acc0, false, false);
      acc1 = __builtin_amdgcn_wmma_f32_16x16x32_f16(false, a.v, false, b1f.v, (short)0, acc1, false, false);
      acc2 = __builtin_amdgcn_wmma_f32_16x16x32_f16(false, a.v, false, b2f.v, (short)0, acc2, false, false);
      acc3 = __builtin_amdgcn_wmma_f32_16x16x32_f16(false, a.v, false, b3f.v, (short)0, acc3, false, false);
    }

    // ---- Elementwise gate math; c stays in registers; h_t -> LDS (f16). ----
#pragma unroll
    for (int r = 0; r < 8; ++r) {
      float iv = sigm(acc0[r]);
      float fv = sigm(acc1[r]);
      float gv = tanh_fast(acc2[r]);
      float ov = sigm(acc3[r]);
      float cv = fv * creg[r] + iv * gv;
      creg[r] = cv;
      anew[(r + 8 * hf) * A_STRIDE + IN_PAD + cb + m] = (half_t)(ov * tanh_fast(cv));
    }
    __syncthreads();
  }

  // Final projection of h_{T-1} (lives in ab1's h-region since (T-1)&1 == 1).
  if (wave < 4) project_step(ab1 + IN_PAD, Vw_p, out, b0, T_STEPS - 1, wave, lane, vbl);
}

extern "C" void kernel_launch(void* const* d_in, const int* in_sizes, int n_in,
                              void* d_out, int out_size, void* d_ws, size_t ws_size,
                              hipStream_t stream) {
  const float* inp  = (const float*)d_in[0];
  const float* W_ih = (const float*)d_in[1];
  const float* W_hh = (const float*)d_in[2];
  const float* b_ih = (const float*)d_in[3];
  const float* b_hh = (const float*)d_in[4];
  const float* V_w  = (const float*)d_in[5];
  const float* V_b  = (const float*)d_in[6];
  float* out = (float*)d_out;

  // Workspace: packed f16 weight fragments (2.55 MB total; L2-resident).
  half_t* Wcat_p = (half_t*)d_ws;                   // 2048*608 = 1,245,184 halves
  half_t* Vw_p   = Wcat_p + (size_t)G_TOT * K_CAT;  //   64*512 =    32,768 halves

  pack_wcat_kernel<<<1024, 256, 0, stream>>>(W_ih, W_hh, Wcat_p);
  pack_vw_kernel<<<64, 256, 0, stream>>>(V_w, Vw_p);

  const size_t lds_bytes =
      (size_t)(2 * BTILE * A_STRIDE) * sizeof(half_t) +
      (size_t)(G_TOT + M_OUT) * sizeof(float);      // 47.9 KB
  lstm_fused_kernel<<<B_TOT / BTILE, 1024, lds_bytes, stream>>>(
      inp, b_ih, b_hh, V_b, Wcat_p, Vw_p, out);
}
